// VQEmbeddingEMA_66116726554649
// MI455X (gfx1250) — matrix operand order repaired and checked by reading, not compiled, not run on hardware
//
#include <hip/hip_runtime.h>
#include <hip/hip_bf16.h>

#define N_TOK   32768   // 32 * 1024 tokens
#define EMB     256
#define K_CODES 1024
#define DECAYF  0.99f

typedef __attribute__((ext_vector_type(16))) __bf16 v16bf;
typedef __attribute__((ext_vector_type(8)))  __bf16 v8bf;
typedef __attribute__((ext_vector_type(8)))  float  v8f;

union AFrag { v16bf v; v8bf h[2]; };

// ---------------------------------------------------------------------------
// 1) Convert codebook fp32 -> bf16 (row major) and compute ||c||^2 per code
//    (on the bf16-rounded values, self-consistent with the WMMA dot products).
//    One wave per code row; 8 waves per block.
// ---------------------------------------------------------------------------
__global__ __launch_bounds__(256) void vq_prep(const float* __restrict__ emb,
                                               __bf16* __restrict__ cbBf,
                                               float* __restrict__ csq) {
    int w    = threadIdx.x >> 5;
    int lane = threadIdx.x & 31;
    int k    = blockIdx.x * 8 + w;           // code row, 0..1023

    const float4* e4 = (const float4*)(emb + (size_t)k * EMB);
    float4 a = e4[lane * 2 + 0];
    float4 b = e4[lane * 2 + 1];

    __bf16 r[8];
    r[0] = (__bf16)a.x; r[1] = (__bf16)a.y; r[2] = (__bf16)a.z; r[3] = (__bf16)a.w;
    r[4] = (__bf16)b.x; r[5] = (__bf16)b.y; r[6] = (__bf16)b.z; r[7] = (__bf16)b.w;

    float s = 0.0f;
    __bf16* dst = cbBf + (size_t)k * EMB + lane * 8;
    #pragma unroll
    for (int i = 0; i < 8; ++i) {
        float f = (float)r[i];
        s += f * f;
        dst[i] = r[i];
    }
    #pragma unroll
    for (int off = 16; off >= 1; off >>= 1) s += __shfl_xor(s, off);
    if (lane == 0) csq[k] = s;
}

// ---------------------------------------------------------------------------
// 2) Zero the scatter accumulators (counts + dw) each call (ws is poisoned).
// ---------------------------------------------------------------------------
__global__ void vq_zero(float* __restrict__ p, int n) {
    int i = blockIdx.x * blockDim.x + threadIdx.x;
    if (i < n) p[i] = 0.0f;
}

// ---------------------------------------------------------------------------
// 3) Nearest-code assignment: fused bf16-WMMA GEMM + argmin.
//    Block = 256 threads = 8 waves. Block stages 128 z rows in LDS (bf16).
//    Each wave owns a 16-row tile; loops 64 column tiles of 16 codes,
//    8x v_wmma_f32_16x16x32_bf16 per tile (D=256).
//    C layout: acc[r] at lane l = row (r + 8*(l>>4)), col (l&15).
// ---------------------------------------------------------------------------
__global__ __launch_bounds__(256)
void vq_assign(const float* __restrict__ z,
               const __bf16* __restrict__ cbBf,
               const float* __restrict__ csq,
               int* __restrict__ idxOut) {
    __shared__ __align__(16) __bf16 sz[128 * EMB];   // 64 KB

    // Stage 128 rows of z, converting fp32 -> bf16.
    const float4* zin = (const float4*)(z + (size_t)blockIdx.x * 128 * EMB);
    for (int i = threadIdx.x; i < 128 * EMB / 4; i += 256) {
        float4 f = zin[i];
        __bf16* d = &sz[i * 4];
        d[0] = (__bf16)f.x; d[1] = (__bf16)f.y;
        d[2] = (__bf16)f.z; d[3] = (__bf16)f.w;
    }
    __syncthreads();

    int w    = threadIdx.x >> 5;
    int lane = threadIdx.x & 31;
    int n    = lane & 15;        // column within tile / A row index
    int hi   = lane >> 4;        // lane half selects K sub-range

    float bestv[8];
    int   besti[8];
    #pragma unroll
    for (int r = 0; r < 8; ++r) { bestv[r] = 3.4e38f; besti[r] = 0; }

    const __bf16* srow = &sz[(size_t)w * 16 * EMB];  // this wave's 16 rows

    for (int ct = 0; ct < K_CODES / 16; ++ct) {
        int   col = ct * 16 + n;
        float cs  = csq[col];
        const __bf16* brow = cbBf + (size_t)col * EMB + hi * 16;

        v8f acc = {};
        #pragma unroll
        for (int kb = 0; kb < EMB; kb += 32) {
            // A fragment: 16 consecutive dims per lane (k = 16*hi + s), matching B.
            AFrag a;
            a.h[0] = *(const v8bf*)(srow + (size_t)n * EMB + kb + hi * 16);
            a.h[1] = *(const v8bf*)(srow + (size_t)n * EMB + kb + hi * 16 + 8);
            // B fragment: 16 consecutive dims of codebook row `col`.
            v16bf bf = *(const v16bf*)(brow + kb);
            acc = __builtin_amdgcn_wmma_f32_16x16x32_bf16(
                false, a.v, false, bf, (short)0, acc, false, false);
        }
        #pragma unroll
        for (int r = 0; r < 8; ++r) {
            float s = cs - 2.0f * acc[r];     // ||c||^2 - 2 x.c
            if (s < bestv[r]) { bestv[r] = s; besti[r] = col; }
        }
    }

    // Min-reduce across the 16 lanes of this half (same rows, disjoint cols).
    #pragma unroll
    for (int r = 0; r < 8; ++r) {
        #pragma unroll
        for (int off = 1; off < 16; off <<= 1) {
            float ov = __shfl_xor(bestv[r], off);
            int   oi = __shfl_xor(besti[r], off);
            if (ov < bestv[r] || (ov == bestv[r] && oi < besti[r])) {
                bestv[r] = ov; besti[r] = oi;
            }
        }
    }
    if (n == 0) {
        int rowBase = blockIdx.x * 128 + w * 16 + hi * 8;
        #pragma unroll
        for (int r = 0; r < 8; ++r) idxOut[rowBase + r] = besti[r];
    }
}

// ---------------------------------------------------------------------------
// 4) Segment counts + segment sum (dw) via fp32 atomics. One wave per token.
// ---------------------------------------------------------------------------
__global__ __launch_bounds__(256)
void vq_scatter(const float* __restrict__ z,
                const int* __restrict__ idx,
                float* __restrict__ counts,
                float* __restrict__ dw) {
    int w    = threadIdx.x >> 5;
    int lane = threadIdx.x & 31;
    int row  = blockIdx.x * 8 + w;
    int c    = idx[row];
    if (lane == 0) atomicAdd(&counts[c], 1.0f);
    const float* zr = z  + (size_t)row * EMB;
    float*       dr = dw + (size_t)c   * EMB;
    #pragma unroll
    for (int j = 0; j < 8; ++j) {
        int d = lane + 32 * j;
        atomicAdd(&dr[d], zr[d]);
    }
}

// ---------------------------------------------------------------------------
// 5) EMA updates -> new_ema_count, new_ema_weight, new_embedding.
// ---------------------------------------------------------------------------
__global__ __launch_bounds__(256)
void vq_finalize(const float* __restrict__ ema_count,
                 const float* __restrict__ ema_weight,
                 const float* __restrict__ counts,
                 const float* __restrict__ dw,
                 float* __restrict__ new_cnt,
                 float* __restrict__ new_w,
                 float* __restrict__ new_emb) {
    int i = blockIdx.x * 256 + threadIdx.x;   // 0 .. K*EMB-1
    int k = i >> 8;
    float nc = DECAYF * ema_count[k] + (1.0f - DECAYF) * counts[k];
    float nw = DECAYF * ema_weight[i] + (1.0f - DECAYF) * dw[i];
    new_w[i]   = nw;
    new_emb[i] = nw / nc;
    if ((i & 255) == 0) new_cnt[k] = nc;
}

// ---------------------------------------------------------------------------
// 6) Gather z_q_x (old codebook) and z_q_x_bar (new codebook), float4 wide.
// ---------------------------------------------------------------------------
__global__ __launch_bounds__(256)
void vq_gather(const int* __restrict__ idx,
               const float* __restrict__ embOld,
               const float* __restrict__ embNew,
               float* __restrict__ zq,
               float* __restrict__ zqb) {
    int i   = blockIdx.x * 256 + threadIdx.x;  // float4 index, N_TOK*64 total
    int row = i >> 6;
    int d4  = i & 63;
    int c   = idx[row];
    const float4* eo = (const float4*)embOld;
    const float4* en = (const float4*)embNew;
    ((float4*)zq)[i]  = eo[c * 64 + d4];
    ((float4*)zqb)[i] = en[c * 64 + d4];
}

extern "C" void kernel_launch(void* const* d_in, const int* in_sizes, int n_in,
                              void* d_out, int out_size, void* d_ws, size_t ws_size,
                              hipStream_t stream) {
    const float* z          = (const float*)d_in[0];   // (32,1024,256)
    const float* emb        = (const float*)d_in[1];   // (1024,256)
    const float* ema_count  = (const float*)d_in[2];   // (1024,)
    const float* ema_weight = (const float*)d_in[3];   // (1024,256)

    float* out       = (float*)d_out;
    float* z_q_x     = out;                    // 8,388,608
    float* z_q_x_bar = out + 8388608;          // 8,388,608
    float* new_emb   = out + 16777216;         // 262,144
    float* new_cnt   = out + 17039360;         // 1,024
    float* new_w     = out + 17040384;         // 262,144

    char* ws = (char*)d_ws;
    __bf16* cbBf  = (__bf16*)(ws + 0);         // 1024*256*2 = 524,288 B
    float*  csq   = (float*)(ws + 524288);     // 4 KB
    int*    idx   = (int*)  (ws + 528384);     // 128 KB
    float*  cnts  = (float*)(ws + 659456);     // 4 KB
    float*  dw    = (float*)(ws + 663552);     // 1 MB
    // cnts and dw are adjacent: zero both with one kernel (263,168 floats)

    vq_prep    <<<K_CODES / 8, 256, 0, stream>>>(emb, cbBf, csq);
    vq_zero    <<<(263168 + 255) / 256, 256, 0, stream>>>(cnts, 263168);
    vq_assign  <<<N_TOK / 128, 256, 0, stream>>>(z, cbBf, csq, idx);
    vq_scatter <<<N_TOK / 8, 256, 0, stream>>>(z, idx, cnts, dw);
    vq_finalize<<<(K_CODES * EMB) / 256, 256, 0, stream>>>(ema_count, ema_weight,
                                                           cnts, dw, new_cnt,
                                                           new_w, new_emb);
    vq_gather  <<<(N_TOK * 64) / 256, 256, 0, stream>>>(idx, emb, new_emb,
                                                        z_q_x, z_q_x_bar);
}
